// LlamaAttentionBlock_19361712571225
// MI455X (gfx1250) — matrix operand order repaired and checked by reading, not compiled, not run on hardware
//
#include <hip/hip_runtime.h>

// ---------------- problem constants (from reference) ----------------
#define BS   8
#define H    32
#define CTX  4096
#define D    128
#define NSPLIT 16
#define KPART (CTX / NSPLIT)   // 256 keys per partition
#define NT   (KPART / 16)      // 16 tiles of 16 keys

typedef _Float16 v16h __attribute__((ext_vector_type(16)));
typedef _Float16 v8h  __attribute__((ext_vector_type(8)));
typedef float    v8f  __attribute__((ext_vector_type(8)));

#define NEG_INF (-__builtin_inff())
#define ROWSTR ((size_t)H * D)   // halves between consecutive keys

__device__ __forceinline__ float readlane_f32(float v, int lane) {
    return __uint_as_float(__builtin_amdgcn_readlane(__float_as_uint(v), lane));
}

// Issue the 8 b128 loads of one 16-key K tile (WMMA A-operand layout).
__device__ __forceinline__
void load_ktile(v8h klo[4], v8h khi[4], const _Float16* __restrict__ krow, int grp)
{
#pragma unroll
    for (int c = 0; c < 4; ++c) {
        const int off = c * 32 + grp * 8;
        klo[c] = *(const v8h*)(krow + off);        // K dims off..off+7
        khi[c] = *(const v8h*)(krow + off + 16);   // K dims off+16..off+23
    }
}

// ---------------------------------------------------------------------
// One 16-key tile, K already resident in registers.
// Order: issue V loads -> WMMA chain -> scalar softmax -> packed-f16 P*V.
// TAIL=true adds xv substitution at key==start and kvlen masking.
// ---------------------------------------------------------------------
template<bool TAIL>
__device__ __forceinline__
void process_tile(const v8h klo[4], const v8h khi[4],
                  int kb, int start, int kvlen, int grp, int lrow,
                  const _Float16* __restrict__ cv_bh,
                  const _Float16* __restrict__ vrow_new,
                  const v16h bq[4],
                  float& m, float& l, v8h& acch)
{
    const float scale = 0.08838834764831845f;  // 1/sqrt(128)

    // ---- (1) V loads first: independent of scores, hide behind WMMA+softmax
    v8h vv[8];
#pragma unroll
    for (int j = 0; j < 8; ++j) {
        const int key = kb + 2 * j + grp;      // group0: even key, group1: odd
        const _Float16* vrow;
        if (TAIL)
            vrow = (key == start) ? vrow_new : cv_bh + (size_t)key * ROWSTR;
        else
            vrow = cv_bh + (size_t)key * ROWSTR;
        vv[j] = *(const v8h*)(vrow + lrow * 8);  // coalesced 16B/lane
    }

    // ---- (2) QK^T: 4x v_wmma_f32_16x16x32_f16 on preloaded K
    v8f cs = {};
#pragma unroll
    for (int c = 0; c < 4; ++c) {
        v16h a = __builtin_shufflevector(klo[c], khi[c],
                 0,1,2,3,4,5,6,7,8,9,10,11,12,13,14,15);
        cs = __builtin_amdgcn_wmma_f32_16x16x32_f16(
                 false, a, false, bq[c], (short)0, cs, false, false);
    }

    // ---- (3) scalar online softmax (scores uniform per 16-lane group)
    float s[8];
    float mt = NEG_INF;
#pragma unroll
    for (int e = 0; e < 8; ++e) {
        float sv = cs[e] * scale;
        if (TAIL) {
            const int key = kb + grp * 8 + e;
            sv = (key < kvlen) ? sv : NEG_INF;
        }
        s[e] = sv;
        mt = fmaxf(mt, sv);
    }
    const float m_new = fmaxf(m, fmaxf(readlane_f32(mt, 0), readlane_f32(mt, 16)));
    const float alpha = __expf(m - m_new);
    {
        const _Float16 ah = (_Float16)alpha;
        const v8h as = {ah, ah, ah, ah, ah, ah, ah, ah};
        acch *= as;                             // 4x v_pk_mul_f16
    }

    float p[8];
    float ls = 0.f;
#pragma unroll
    for (int e = 0; e < 8; ++e) { p[e] = __expf(s[e] - m_new); ls += p[e]; }
    l = l * alpha + readlane_f32(ls, 0) + readlane_f32(ls, 16);
    m = m_new;

    // ---- (4) P*V in packed f16: 4x v_pk_fma_f16 per pair
#pragma unroll
    for (int j = 0; j < 8; ++j) {
        const int ke = 2 * j, ko = 2 * j + 1;
        const float p0 = readlane_f32(p[ke & 7], (ke >> 3) * 16);  // SGPR
        const float p1 = readlane_f32(p[ko & 7], (ko >> 3) * 16);  // SGPR
        const _Float16 ph = (_Float16)(grp ? p1 : p0);
        const v8h ps = {ph, ph, ph, ph, ph, ph, ph, ph};
        acch = vv[j] * ps + acch;
    }
}

// =====================================================================
// Kernel 1: flash-decode partials. One wave32 per (b, h, partition).
// Software-pipelined: K tile t+1 is in flight while tile t computes.
// =====================================================================
__global__ __launch_bounds__(32)
void llama_attn_partial(const _Float16* __restrict__ xq,
                        const _Float16* __restrict__ xk,
                        const _Float16* __restrict__ xv,
                        const _Float16* __restrict__ cache_k,
                        const _Float16* __restrict__ cache_v,
                        const int* __restrict__ start_index_p,
                        float* __restrict__ pacc,   // [BS*H*NSPLIT][D]
                        float* __restrict__ pm,     // [BS*H*NSPLIT]
                        float* __restrict__ pl)     // [BS*H*NSPLIT]
{
    const int lane  = threadIdx.x;            // 0..31 (wave32)
    const int blk   = blockIdx.x;             // ((b*H)+h)*NSPLIT + part
    const int part  = blk % NSPLIT;
    const int h     = (blk / NSPLIT) % H;
    const int b     = blk / (NSPLIT * H);
    const int start = *start_index_p;         // 4095
    const int kvlen = start + 1;

    const int grp  = lane >> 4;               // lane group 0 / 1
    const int lrow = lane & 15;

    // q in WMMA-B layout: lanes 0-15 carry K=c*32+{0..7,16..23}, 16-31 K=c*32+{8..15,24..31}
    const _Float16* qrow = xq + ((size_t)b * H + h) * (size_t)D;
    v16h bq[4];
#pragma unroll
    for (int c = 0; c < 4; ++c)
        bq[c] = __builtin_shufflevector(*(const v8h*)(qrow + c * 32 + grp * 8), *(const v8h*)(qrow + c * 32 + grp * 8 + 16), 0,1,2,3,4,5,6,7,8,9,10,11,12,13,14,15);

    const _Float16* krow_new = xk + ((size_t)b * H + h) * (size_t)D;
    const _Float16* vrow_new = xv + ((size_t)b * H + h) * (size_t)D;
    const _Float16* ck_bh = cache_k + ((size_t)b * CTX * H + h) * (size_t)D;
    const _Float16* cv_bh = cache_v + ((size_t)b * CTX * H + h) * (size_t)D;

    float m = NEG_INF;
    float l = 0.0f;
    v8h acch = {};                             // packed-f16 P*V accumulator

    const int k0 = part * KPART;

    // K-row pointer with tail-aware xk substitution (uniform outer branch)
    auto krow_ptr = [&](int kb) -> const _Float16* {
        const int mykey = kb + lrow;
        if (kb + 16 > start)                   // wave-uniform
            return (mykey == start) ? krow_new : ck_bh + (size_t)mykey * ROWSTR;
        return ck_bh + (size_t)mykey * ROWSTR;
    };

    // ---- prologue: K tile 0 in flight
    v8h klo[2][4], khi[2][4];
    load_ktile(klo[0], khi[0], krow_ptr(k0), grp);

#pragma unroll 2
    for (int t = 0; t < NT; ++t) {
        const int kb  = k0 + t * 16;
        const int cur = t & 1, nxt = cur ^ 1;

        // prefetch next K tile before consuming anything of this tile
        if (t + 1 < NT)
            load_ktile(klo[nxt], khi[nxt], krow_ptr(kb + 16), grp);

        if (kb + 16 <= start) {        // wave-uniform; EXEC stays full
            process_tile<false>(klo[cur], khi[cur], kb, start, kvlen, grp, lrow,
                                cv_bh, vrow_new, bq, m, l, acch);
        } else {
            process_tile<true>(klo[cur], khi[cur], kb, start, kvlen, grp, lrow,
                               cv_bh, vrow_new, bq, m, l, acch);
        }
    }

    // widen once, then merge the two lane groups' (even/odd-key) partials
    float accf[8];
#pragma unroll
    for (int i = 0; i < 8; ++i) accf[i] = (float)acch[i];
#pragma unroll
    for (int i = 0; i < 8; ++i) accf[i] += __shfl_xor(accf[i], 16, 32);

    // write partial (acc, m, l); lanes 0-15 own dims [lrow*8, lrow*8+8)
    if (lane < 16) {
        float* outp = pacc + (size_t)blk * D + lrow * 8;
#pragma unroll
        for (int i = 0; i < 8; ++i) outp[i] = accf[i];
    }
    if (lane == 0) { pm[blk] = m; pl[blk] = l; }
}

// =====================================================================
// Kernel 2: combine NSPLIT partials per (b,h); emit f16 output.
// =====================================================================
__global__ __launch_bounds__(128)
void llama_attn_reduce(const float* __restrict__ pacc,
                       const float* __restrict__ pm,
                       const float* __restrict__ pl,
                       _Float16* __restrict__ out)   // [BS][H*D]
{
    const int d    = threadIdx.x;   // 0..127
    const int bh   = blockIdx.x;    // b*H + h
    const int base = bh * NSPLIT;

    float mg = NEG_INF;
#pragma unroll
    for (int p = 0; p < NSPLIT; ++p) mg = fmaxf(mg, pm[base + p]);

    float lg = 0.f, og = 0.f;
#pragma unroll
    for (int p = 0; p < NSPLIT; ++p) {
        const float w = __expf(pm[base + p] - mg);
        lg += w * pl[base + p];
        og += w * pacc[(size_t)(base + p) * D + d];
    }
    out[(size_t)bh * D + d] = (_Float16)(og / lg);
}

// =====================================================================
extern "C" void kernel_launch(void* const* d_in, const int* in_sizes, int n_in,
                              void* d_out, int out_size, void* d_ws, size_t ws_size,
                              hipStream_t stream)
{
    const _Float16* xq      = (const _Float16*)d_in[0];
    const _Float16* xk      = (const _Float16*)d_in[1];
    const _Float16* xv      = (const _Float16*)d_in[2];
    const _Float16* cache_k = (const _Float16*)d_in[3];
    const _Float16* cache_v = (const _Float16*)d_in[4];
    const int*      start   = (const int*)d_in[5];

    const size_t nparts = (size_t)BS * H * NSPLIT;        // 4096
    float* pacc = (float*)d_ws;                           // nparts * D floats
    float* pm   = pacc + nparts * D;                      // nparts floats
    float* pl   = pm + nparts;                            // nparts floats

    llama_attn_partial<<<dim3((unsigned)nparts), dim3(32), 0, stream>>>(
        xq, xk, xv, cache_k, cache_v, start, pacc, pm, pl);

    llama_attn_reduce<<<dim3(BS * H), dim3(128), 0, stream>>>(
        pacc, pm, pl, (_Float16*)d_out);
}
